// HModelEncoder_5111011083055
// MI455X (gfx1250) — matrix operand order repaired and checked by reading, not compiled, not run on hardware
//
#include <hip/hip_runtime.h>
#include <hip/hip_bf16.h>

// ---------------- problem constants ----------------
#define NN   32768          // nodes
#define KK   8              // mail per node
#define EE   (NN * KK)      // edges = 262144
#define DD   300            // feature dim
#define NP   320            // padded feature dim (mult of 32 for WMMA K-step)
#define HH   4              // heads
#define DKH  75             // dim per head
#define KP2  960            // padded 3*DD for final GEMM

typedef __attribute__((ext_vector_type(16))) __bf16 v16bf;
typedef __attribute__((ext_vector_type(8)))  float  v8f;

// ---------------- bf16 helpers (round-to-nearest-even) ----------------
__device__ __forceinline__ unsigned short f2bf(float f) {
    union { float f; unsigned u; } x; x.f = f;
    unsigned r = x.u + 0x7FFFu + ((x.u >> 16) & 1u);
    return (unsigned short)(r >> 16);
}
__device__ __forceinline__ float bf2f(unsigned short b) {
    union { unsigned u; float f; } x; x.u = ((unsigned)b) << 16; return x.f;
}

// ---------------- weight convert + transpose (W[k][n] fp32 -> WT[n][k] bf16, zero padded) ----------------
__global__ void wtpose(const float* __restrict__ W, unsigned short* __restrict__ WT,
                       int Krows, int Ncols, int Kp, int Np) {
    int idx = blockIdx.x * blockDim.x + threadIdx.x;
    if (idx >= Kp * Np) return;
    int n = idx / Kp, k = idx % Kp;
    float v = (n < Ncols && k < Krows) ? W[(size_t)k * Ncols + n] : 0.f;
    WT[idx] = f2bf(v);
}

// ---------------- fp32 (rows x 300) -> bf16 (rows x 320, zero padded) ----------------
__global__ void rowcvt(const float* __restrict__ S, unsigned short* __restrict__ Dst) {
    size_t r = blockIdx.x;
    for (int c = threadIdx.x; c < NP; c += blockDim.x)
        Dst[r * NP + c] = (c < DD) ? f2bf(S[r * DD + c]) : (unsigned short)0;
}

// ---------------- mail gather: mail = h[mail_idx], mailpf = mail + f_h[node] ----------------
__global__ void gather_mail(const unsigned short* __restrict__ h,
                            const unsigned short* __restrict__ fh,
                            const int* __restrict__ midx,
                            unsigned short* __restrict__ mail,
                            unsigned short* __restrict__ mailpf) {
    int e = blockIdx.x;
    int n = e >> 3;
    size_t s = (size_t)midx[e] * NP;
    size_t o = (size_t)e * NP;
    size_t fo = (size_t)n * NP;
    for (int c = threadIdx.x; c < NP; c += blockDim.x) {
        unsigned short m = h[s + c];
        mail[o + c]   = m;
        mailpf[o + c] = f2bf(bf2f(m) + bf2f(fh[fo + c]));
    }
}

// ---------------- m = f_h[src_idx[e]] - h[e^1] ----------------
__global__ void build_m(const unsigned short* __restrict__ fh,
                        const unsigned short* __restrict__ h,
                        const int* __restrict__ sidx,
                        unsigned short* __restrict__ m) {
    int e = blockIdx.x;
    size_t s = (size_t)sidx[e] * NP;
    size_t r = (size_t)(e ^ 1) * NP;
    size_t o = (size_t)e * NP;
    for (int c = threadIdx.x; c < NP; c += blockDim.x)
        m[o + c] = f2bf(bf2f(fh[s + c]) - bf2f(h[r + c]));
}

// ---------------- per-(node,head) 8-key softmax attention; one wave each ----------------
__global__ void attn(const float* __restrict__ Q, const float* __restrict__ Km,
                     const float* __restrict__ Vm, unsigned short* __restrict__ O) {
    int wave = threadIdx.x >> 5, lane = threadIdx.x & 31;
    int g    = blockIdx.x * 8 + wave;
    int node = g >> 2, head = g & 3;
    const float* q = Q + (size_t)node * NP + head * DKH;
    float s[KK];
    #pragma unroll
    for (int j = 0; j < KK; ++j) {
        const float* kp = Km + (size_t)(node * KK + j) * NP + head * DKH;
        float p = 0.f;
        for (int d = lane; d < DKH; d += 32) p += q[d] * kp[d];
        #pragma unroll
        for (int off = 16; off; off >>= 1) p += __shfl_xor(p, off, 32);
        s[j] = p * 0.11547005383792515f;  // 1/sqrt(75)
    }
    float mx = s[0];
    #pragma unroll
    for (int j = 1; j < KK; ++j) mx = fmaxf(mx, s[j]);
    float sum = 0.f;
    #pragma unroll
    for (int j = 0; j < KK; ++j) { s[j] = __expf(s[j] - mx); sum += s[j]; }
    float inv = 1.f / sum;
    for (int d = lane; d < DKH; d += 32) {
        float o = 0.f;
        #pragma unroll
        for (int j = 0; j < KK; ++j)
            o += s[j] * inv * Vm[(size_t)(node * KK + j) * NP + head * DKH + d];
        O[(size_t)node * NP + head * DKH + d] = f2bf(o);
    }
    if (head == 0)
        for (int c = DD + lane; c < NP; c += 32) O[(size_t)node * NP + c] = 0;
}

// ---------------- concat(mail_sum, f_h, f) -> bf16 (N x 960, zero padded) ----------------
__global__ void concat(const unsigned short* __restrict__ h,
                       const unsigned short* __restrict__ fh,
                       const float* __restrict__ f,
                       const int* __restrict__ midx,
                       unsigned short* __restrict__ cc) {
    int n = blockIdx.x;
    for (int c = threadIdx.x; c < KP2; c += blockDim.x) {
        unsigned short v;
        if (c < DD) {
            float s = 0.f;
            #pragma unroll
            for (int j = 0; j < KK; ++j)
                s += bf2f(h[(size_t)midx[n * KK + j] * NP + c]);
            v = f2bf(s);
        } else if (c < 2 * DD) {
            v = fh[(size_t)n * NP + (c - DD)];
        } else if (c < 3 * DD) {
            v = f2bf(f[(size_t)n * DD + (c - 2 * DD)]);
        } else v = 0;
        cc[(size_t)n * KP2 + c] = v;
    }
}

// ---------------- tiled bf16 WMMA GEMM with async global->LDS double buffering ----------------
// C = A(bf16, M x Kp) * Bt(bf16, [Np x Kp] = col-major B) + bias
// block = 256 threads (8 waves), workgroup tile 128(M) x 64(N), wave tile 32x32 (2x2 WMMA)
// LDS: 2 buffers x (A 128x32 + B 64x32) bf16 = 2 x 12 KB; staged with
// GLOBAL_LOAD_ASYNC_TO_LDS_B128 (ASYNCcnt), one s_wait_asynccnt + one barrier per K step.
// epi: 0 = fp32 padded (ldc=NP); 1 = bf16 padded; 2 = relu(x + .) -> bf16 padded; 3 = fp32 unpadded final
#define TILE_US 6144   // ushorts per LDS buffer (A 4096 + B 2048)

__global__ __launch_bounds__(256)
void gemm_bf16(const unsigned short* __restrict__ A, int lda,
               const unsigned short* __restrict__ Bt, int ldb,
               const float* __restrict__ bias,
               float* __restrict__ Cf, unsigned short* __restrict__ Cb,
               const float* __restrict__ Xadd, float* __restrict__ Of,
               int Kp, int epi) {
    __shared__ __align__(16) unsigned short smem[2 * TILE_US];
    const int tid  = threadIdx.x;
    const int m0   = blockIdx.x * 128, n0 = blockIdx.y * 64;
    const int wave = tid >> 5, lane = tid & 31;
    const int wr   = wave & 3, wc = wave >> 2;      // 4x2 wave grid
    const int lrow = lane & 15, lhalf = lane >> 4;

    // wave-relative LDS byte offset of smem (FLAT->LDS mapping uses addr[31:0])
    const unsigned ldsbase = (unsigned)(unsigned long long)(&smem[0]);
    const int br = tid >> 2, bs = tid & 3;          // B tile: 64 rows x 4 b128 segs

    // issue async loads for one 128x32 A tile + 64x32 B tile into buffer `buf`
    auto issue = [&](int k0, int buf) {
        unsigned base = ldsbase + (unsigned)buf * (TILE_US * 2);
        {   // B segment (1 per thread)
            unsigned dst = base + 8192u + (unsigned)(br * 32 + bs * 8) * 2u;
            const unsigned short* src = Bt + (size_t)(n0 + br) * ldb + k0 + bs * 8;
            asm volatile("global_load_async_to_lds_b128 %0, %1, off"
                         :: "v"(dst), "v"(src) : "memory");
        }
        #pragma unroll
        for (int t = 0; t < 2; ++t) {   // A segments (2 per thread)
            int uu = tid + t * 256;
            int ar = uu >> 2, as = uu & 3;
            unsigned dst = base + (unsigned)(ar * 32 + as * 8) * 2u;
            const unsigned short* src = A + (size_t)(m0 + ar) * lda + k0 + as * 8;
            asm volatile("global_load_async_to_lds_b128 %0, %1, off"
                         :: "v"(dst), "v"(src) : "memory");
        }
    };

    v8f acc[2][2] = {};
    const int nk = Kp >> 5;

    issue(0, 0);
    for (int kt = 0; kt < nk; ++kt) {
        const int cur = kt & 1;
        asm volatile("s_wait_asynccnt 0x0" ::: "memory");  // this wave's tile-kt loads done
        __syncthreads();                                   // all waves' tiles visible; prev buf free
        if (kt + 1 < nk) issue((kt + 1) << 5, 1 - cur);

        const unsigned short* sA = &smem[cur * TILE_US];
        const unsigned short* sB = &smem[cur * TILE_US + 4096];

        union FB { unsigned u[8]; v16bf v; };
        FB afr[2], bfr[2];
        #pragma unroll
        for (int tm = 0; tm < 2; ++tm) {            // A 16x32 bf16 fragment (ISA 7.12.2 layout)
            int row = wr * 32 + tm * 16 + lrow;
            #pragma unroll
            for (int vv = 0; vv < 8; ++vv) {
                int k = (vv < 4 ? 2 * vv : 16 + 2 * (vv - 4)) + lhalf * 8;
                afr[tm].u[vv] = *(const unsigned*)(&sA[row * 32 + k]);
            }
        }
        #pragma unroll
        for (int tn = 0; tn < 2; ++tn) {            // B 32x16 bf16 fragment
            int col = wc * 32 + tn * 16 + lrow;
            #pragma unroll
            for (int vv = 0; vv < 8; ++vv) {
                int k = (vv < 4 ? 2 * vv : 16 + 2 * (vv - 4)) + lhalf * 8;
                bfr[tn].u[vv] = *(const unsigned*)(&sB[col * 32 + k]);
            }
        }
        #pragma unroll
        for (int tm = 0; tm < 2; ++tm)
            #pragma unroll
            for (int tn = 0; tn < 2; ++tn)
                acc[tm][tn] = __builtin_amdgcn_wmma_f32_16x16x32_bf16(
                    false, afr[tm].v, false, bfr[tn].v,
                    (short)0, acc[tm][tn], false, false);
    }

    // epilogue (C/D layout: VGPR i -> M=i (lanes 0-15) / M=8+i (lanes 16-31), N=lane%16)
    #pragma unroll
    for (int tm = 0; tm < 2; ++tm)
        #pragma unroll
        for (int tn = 0; tn < 2; ++tn)
            #pragma unroll
            for (int i = 0; i < 8; ++i) {
                int row = m0 + wr * 32 + tm * 16 + lhalf * 8 + i;
                int col = n0 + wc * 32 + tn * 16 + lrow;
                float v = acc[tm][tn][i] + (col < DD ? bias[col] : 0.f);
                if (epi == 0) {
                    Cf[(size_t)row * NP + col] = v;
                } else if (epi == 1) {
                    Cb[(size_t)row * NP + col] = f2bf(v);
                } else if (epi == 2) {
                    if (col < DD) v += Xadd[(size_t)row * DD + col];
                    v = fmaxf(v, 0.f);
                    Cb[(size_t)row * NP + col] = f2bf(v);
                } else {
                    if (col < DD) Of[(size_t)row * DD + col] = v;
                }
            }
}

// ---------------- host-side orchestration ----------------
extern "C" void kernel_launch(void* const* d_in, const int* in_sizes, int n_in,
                              void* d_out, int out_size, void* d_ws, size_t ws_size,
                              hipStream_t stream) {
    (void)in_sizes; (void)n_in; (void)out_size; (void)ws_size;
    const float* f     = (const float*)d_in[0];
    const float* x     = (const float*)d_in[1];
    const int*   midx  = (const int*)d_in[2];
    const int*   sidx  = (const int*)d_in[3];
    const float* Wq    = (const float*)d_in[4];  const float* bq    = (const float*)d_in[5];
    const float* Wk    = (const float*)d_in[6];  const float* bk    = (const float*)d_in[7];
    const float* Wv    = (const float*)d_in[8];  const float* bv    = (const float*)d_in[9];
    const float* Wo    = (const float*)d_in[10]; const float* bo    = (const float*)d_in[11];
    const float* Wmp   = (const float*)d_in[12]; const float* bmp   = (const float*)d_in[13];
    const float* Wlast = (const float*)d_in[14]; const float* blast = (const float*)d_in[15];
    float* out = (float*)d_out;

    char* ws = (char*)d_ws;
    size_t off = 0;
    auto alloc = [&](size_t bytes) -> void* {
        void* p = ws + off;
        off = (off + bytes + 255) & ~(size_t)255;
        return p;
    };

    unsigned short* wqT  = (unsigned short*)alloc((size_t)NP * NP * 2);
    unsigned short* wkT  = (unsigned short*)alloc((size_t)NP * NP * 2);
    unsigned short* wvT  = (unsigned short*)alloc((size_t)NP * NP * 2);
    unsigned short* woT  = (unsigned short*)alloc((size_t)NP * NP * 2);
    unsigned short* wm0T = (unsigned short*)alloc((size_t)NP * NP * 2);
    unsigned short* wm1T = (unsigned short*)alloc((size_t)NP * NP * 2);
    unsigned short* wlT  = (unsigned short*)alloc((size_t)NP * KP2 * 2);
    unsigned short* fbf  = (unsigned short*)alloc((size_t)NN * NP * 2);
    unsigned short* fhbf = (unsigned short*)alloc((size_t)NN * NP * 2);
    unsigned short* hbf  = (unsigned short*)alloc((size_t)EE * NP * 2);
    unsigned short* mail = (unsigned short*)alloc((size_t)EE * NP * 2);  // reused for m
    unsigned short* mpf  = (unsigned short*)alloc((size_t)EE * NP * 2);
    float*          Qf   = (float*)alloc((size_t)NN * NP * 4);
    float*          Kf   = (float*)alloc((size_t)EE * NP * 4);
    float*          Vf   = (float*)alloc((size_t)EE * NP * 4);
    unsigned short* obf  = (unsigned short*)alloc((size_t)NN * NP * 2);
    unsigned short* ccbf = (unsigned short*)alloc((size_t)NN * KP2 * 2);

    dim3 b256(256);
    int wblk = (NP * NP + 255) / 256;
    wtpose<<<wblk, b256, 0, stream>>>(Wq, wqT, DD, DD, NP, NP);
    wtpose<<<wblk, b256, 0, stream>>>(Wk, wkT, DD, DD, NP, NP);
    wtpose<<<wblk, b256, 0, stream>>>(Wv, wvT, DD, DD, NP, NP);
    wtpose<<<wblk, b256, 0, stream>>>(Wo, woT, DD, DD, NP, NP);
    wtpose<<<wblk, b256, 0, stream>>>(Wmp,            wm0T, DD, DD, NP, NP);
    wtpose<<<wblk, b256, 0, stream>>>(Wmp + DD * DD,  wm1T, DD, DD, NP, NP);
    wtpose<<<(NP * KP2 + 255) / 256, b256, 0, stream>>>(Wlast, wlT, 3 * DD, DD, KP2, NP);

    rowcvt<<<NN, b256, 0, stream>>>(f, fbf);
    rowcvt<<<NN, b256, 0, stream>>>(f, fhbf);
    rowcvt<<<EE, b256, 0, stream>>>(x, hbf);

    unsigned short* wmT[2] = {wm0T, wm1T};
    dim3 gN(NN / 128, NP / 64);
    dim3 gE(EE / 128, NP / 64);
    for (int i = 0; i < 2; ++i) {
        gather_mail<<<EE, b256, 0, stream>>>(hbf, fhbf, midx, mail, mpf);
        gemm_bf16<<<gN, b256, 0, stream>>>(fhbf, NP, wqT, NP, bq, Qf, nullptr, nullptr, nullptr, NP, 0);
        gemm_bf16<<<gE, b256, 0, stream>>>(mail, NP, wkT, NP, bk, Kf, nullptr, nullptr, nullptr, NP, 0);
        gemm_bf16<<<gE, b256, 0, stream>>>(mpf,  NP, wvT, NP, bv, Vf, nullptr, nullptr, nullptr, NP, 0);
        attn<<<NN * HH / 8, b256, 0, stream>>>(Qf, Kf, Vf, obf);
        gemm_bf16<<<gN, b256, 0, stream>>>(obf, NP, woT, NP, bo, nullptr, fhbf, nullptr, nullptr, NP, 1);
        build_m<<<EE, b256, 0, stream>>>(fhbf, hbf, sidx, mail);
        gemm_bf16<<<gE, b256, 0, stream>>>(mail, NP, wmT[i], NP, bmp + i * DD, nullptr, hbf, x, nullptr, NP, 2);
    }
    concat<<<NN, b256, 0, stream>>>(hbf, fhbf, f, midx, ccbf);
    gemm_bf16<<<gN, b256, 0, stream>>>(ccbf, KP2, wlT, KP2, blast, nullptr, nullptr, nullptr, out, KP2, 3);
}
